// TopKRouter_19104014532973
// MI455X (gfx1250) — matrix operand order repaired and checked by reading, not compiled, not run on hardware
//
#include <hip/hip_runtime.h>
#include <hip/hip_bf16.h>

#define N_TOKENS 16384
#define HID      2048
#define N_EXP    64
#define TOPK     8
#define KSPLIT   4
#define KCHUNK   (HID / KSPLIT)               // 512
#define KSTEPS_W (KCHUNK / 32)                // 16 wmma K-steps per wave
#define NTILES   (N_TOKENS / 16)              // 1024
#define TOK_PER_BLOCK 128                     // 8 waves x 16 tokens (post kernel)
#define NPBLOCKS (N_TOKENS / TOK_PER_BLOCK)   // 128

typedef __attribute__((ext_vector_type(8)))  float  v8f;
typedef __attribute__((ext_vector_type(16))) __bf16 v16bf;

__device__ __forceinline__ unsigned short f2bf_rne(float f) {
    unsigned int u = __builtin_bit_cast(unsigned int, f);
    unsigned int r = u + 0x7FFFu + ((u >> 16) & 1u);
    return (unsigned short)(r >> 16);
}

// ---------------------------------------------------------------------------
// Kernel 1: pack gate_w (2048 x 64 fp32) into the 16-bit B-matrix (32x16)
// WMMA lane layout: lane = hb*16 + n, element e = B[e<8 ? hb*8+e : 16+hb*8+(e-8)][n].
// packed[((kkg*4 + t)*32 + lane)*16 + e] = gate_w[kkg*32 + (e<8 ? hb*8+e : 16+hb*8+(e-8))][t*16 + n]
// ---------------------------------------------------------------------------
__global__ __launch_bounds__(256) void pack_b_kernel(
    const float* __restrict__ gw, unsigned short* __restrict__ pb)
{
    int idx = blockIdx.x * blockDim.x + threadIdx.x;    // 0 .. 8191
    if (idx >= (HID / 32) * 4 * 32) return;
    int lane = idx & 31;
    int t    = (idx >> 5) & 3;
    int kkg  = idx >> 7;
    int n    = lane & 15;
    int hb   = lane >> 4;
    unsigned short* dst = pb + (size_t)idx * 16;
#pragma unroll
    for (int e = 0; e < 16; ++e) {
        int k = kkg * 32 + (e < 8 ? hb * 8 + e : 16 + hb * 8 + (e - 8));
        dst[e] = f2bf_rne(gw[(size_t)k * N_EXP + t * 16 + n]);
    }
}

// ---------------------------------------------------------------------------
// Kernel 2: streaming GEMM. One wave per (16-token tile, K-quarter).
// 4096 waves total. Double-buffered loads -> wmma never waits on loadcnt 0.
// Partial logits: part[((tile*4 + kp)*16 + row)*64 + e]
// ---------------------------------------------------------------------------
__global__ __launch_bounds__(256) void gemm_kernel(
    const float* __restrict__ hidden, const unsigned short* __restrict__ pb,
    float* __restrict__ part)
{
    const int tid  = threadIdx.x;
    const int wave = tid >> 5;
    const int lane = tid & 31;
    const int kp   = blockIdx.x & (KSPLIT - 1);
    const int tile = (blockIdx.x >> 2) * 8 + wave;      // 0..1023
    const int m    = lane & 15;
    const int ha   = lane >> 4;

    const float* __restrict__ arow =
        hidden + (size_t)(tile * 16 + m) * HID + (size_t)kp * KCHUNK;
    const v16bf* __restrict__ bpbase =
        (const v16bf*)pb + (size_t)kp * (KSTEPS_W * 4 * 32) + lane;

    v8f acc0 = {}, acc1 = {}, acc2 = {}, acc3 = {};

    // prologue: loads for kk = 0
    const float4* p0 = (const float4*)(arow + ha * 8);
    const float4* p1 = (const float4*)(arow + 16 + ha * 8);
    float4 nx0 = p0[0], nx1 = p0[1], ny0 = p1[0], ny1 = p1[1];
    v16bf nb0 = bpbase[0], nb1 = bpbase[32], nb2 = bpbase[64], nb3 = bpbase[96];

    for (int kk = 0; kk < KSTEPS_W; ++kk) {
        float4 x0 = nx0, x1 = nx1, y0 = ny0, y1 = ny1;
        v16bf b0 = nb0, b1 = nb1, b2 = nb2, b3 = nb3;

        if (kk + 1 < KSTEPS_W) {                // issue next-iteration loads NOW
            const float* an = arow + (kk + 1) * 32;
            const float4* q0 = (const float4*)(an + ha * 8);
            const float4* q1 = (const float4*)(an + 16 + ha * 8);
            nx0 = q0[0]; nx1 = q0[1]; ny0 = q1[0]; ny1 = q1[1];
            const v16bf* bn = bpbase + (size_t)(kk + 1) * 128;
            nb0 = bn[0]; nb1 = bn[32]; nb2 = bn[64]; nb3 = bn[96];
            __builtin_prefetch(an + 64, 0, 0);  // HBM stream-ahead
        }

        // native f32 -> bf16 converts (hardware cvt, RNE)
        union { v16bf v; __bf16 e[16]; } a;
        a.e[0]  = (__bf16)x0.x; a.e[1]  = (__bf16)x0.y;
        a.e[2]  = (__bf16)x0.z; a.e[3]  = (__bf16)x0.w;
        a.e[4]  = (__bf16)x1.x; a.e[5]  = (__bf16)x1.y;
        a.e[6]  = (__bf16)x1.z; a.e[7]  = (__bf16)x1.w;
        a.e[8]  = (__bf16)y0.x; a.e[9]  = (__bf16)y0.y;
        a.e[10] = (__bf16)y0.z; a.e[11] = (__bf16)y0.w;
        a.e[12] = (__bf16)y1.x; a.e[13] = (__bf16)y1.y;
        a.e[14] = (__bf16)y1.z; a.e[15] = (__bf16)y1.w;

        acc0 = __builtin_amdgcn_wmma_f32_16x16x32_bf16(false, a.v, false, b0, (short)0, acc0, false, false);
        acc1 = __builtin_amdgcn_wmma_f32_16x16x32_bf16(false, a.v, false, b1, (short)0, acc1, false, false);
        acc2 = __builtin_amdgcn_wmma_f32_16x16x32_bf16(false, a.v, false, b2, (short)0, acc2, false, false);
        acc3 = __builtin_amdgcn_wmma_f32_16x16x32_bf16(false, a.v, false, b3, (short)0, acc3, false, false);
    }

    // store partial tile: C/D layout lane = hc*16+n, VGPR r -> row hc*8+r, col n
    const int n = lane & 15;
    float* dst = part + ((size_t)(tile * KSPLIT + kp) * 16) * N_EXP;
#pragma unroll
    for (int r = 0; r < 8; ++r) {
        const int row = ha * 8 + r;
        dst[(size_t)row * N_EXP +  0 + n] = acc0[r];
        dst[(size_t)row * N_EXP + 16 + n] = acc1[r];
        dst[(size_t)row * N_EXP + 32 + n] = acc2[r];
        dst[(size_t)row * N_EXP + 48 + n] = acc3[r];
    }
}

// ---------------------------------------------------------------------------
// Kernel 3: sum K-partials, top-8 + softmax, aux-loss partials.
// ---------------------------------------------------------------------------
__global__ __launch_bounds__(256) void postproc_kernel(
    const float* __restrict__ part,
    float* __restrict__ out_w, float* __restrict__ out_i,
    float* __restrict__ ws_prob, float* __restrict__ ws_cnt)
{
    __shared__ float s_log[TOK_PER_BLOCK][N_EXP + 4];
    __shared__ unsigned int s_icnt[N_EXP];

    const int tid = threadIdx.x;
    if (tid < N_EXP) s_icnt[tid] = 0u;

    // gather + reduce the 4 K-partials (fixed order -> deterministic)
    for (int g = tid; g < TOK_PER_BLOCK * 16; g += 256) {
        const int j    = g >> 4;            // token within block
        const int e4   = (g & 15) * 4;      // expert group of 4
        const int tile = blockIdx.x * 8 + (j >> 4);
        const int row  = j & 15;
        const float4* q = (const float4*)(part +
            ((size_t)(tile * KSPLIT) * 16 + row) * N_EXP + e4);
        // kp stride = 16*64 floats = 256 float4
        float4 s0 = q[0], s1 = q[256], s2 = q[512], s3 = q[768];
        s_log[j][e4 + 0] = s0.x + s1.x + s2.x + s3.x;
        s_log[j][e4 + 1] = s0.y + s1.y + s2.y + s3.y;
        s_log[j][e4 + 2] = s0.z + s1.z + s2.z + s3.z;
        s_log[j][e4 + 3] = s0.w + s1.w + s2.w + s3.w;
    }
    __syncthreads();

    if (tid < TOK_PER_BLOCK) {
        const int j = tid;
        const size_t gtok = (size_t)blockIdx.x * TOK_PER_BLOCK + j;

        float tv[TOPK]; int tix[TOPK];
#pragma unroll
        for (int s = 0; s < TOPK; ++s) { tv[s] = -3.4e38f; tix[s] = 0; }

        for (int e = 0; e < N_EXP; ++e) {   // unrolled compare-exchange insert
            float v = s_log[j][e]; int vi = e;
#pragma unroll
            for (int s = 0; s < TOPK; ++s) {
                bool gt = v > tv[s];
                float nv = gt ? v : tv[s];  int ni = gt ? vi : tix[s];
                float ov = gt ? tv[s] : v;  int oi = gt ? tix[s] : vi;
                tv[s] = nv; tix[s] = ni; v = ov; vi = oi;
            }
        }

        const float mx = tv[0];
        float w[TOPK]; float se = 0.f;
#pragma unroll
        for (int s = 0; s < TOPK; ++s) { w[s] = __expf(tv[s] - mx); se += w[s]; }
        const float inv = 1.f / se;
#pragma unroll
        for (int s = 0; s < TOPK; ++s) {
            out_w[gtok * TOPK + s] = w[s] * inv;
            out_i[gtok * TOPK + s] = (float)tix[s];
            atomicAdd(&s_icnt[tix[s]], 1u);          // integer: exact, deterministic
        }

        float sa = 0.f;
        for (int e = 0; e < N_EXP; ++e) sa += __expf(s_log[j][e] - mx);
        const float isa = 1.f / sa;
        for (int e = 0; e < N_EXP; ++e)
            s_log[j][e] = __expf(s_log[j][e] - mx) * isa;   // logits -> probs
    }
    __syncthreads();

    if (tid < N_EXP) {                                      // deterministic column sums
        float s = 0.f;
        for (int j = 0; j < TOK_PER_BLOCK; ++j) s += s_log[j][tid];
        ws_prob[(size_t)blockIdx.x * N_EXP + tid] = s;
        ws_cnt [(size_t)blockIdx.x * N_EXP + tid] = (float)s_icnt[tid];
    }
}

// ---------------------------------------------------------------------------
// Kernel 4: deterministic final reduction -> aux loss scalar
// ---------------------------------------------------------------------------
__global__ __launch_bounds__(64) void aux_kernel(
    const float* __restrict__ ws_prob, const float* __restrict__ ws_cnt,
    float* __restrict__ out_aux)
{
    __shared__ float red[N_EXP];
    const int e = threadIdx.x;
    float p = 0.f, c = 0.f;
    for (int b = 0; b < NPBLOCKS; ++b) {
        p += ws_prob[(size_t)b * N_EXP + e];
        c += ws_cnt [(size_t)b * N_EXP + e];
    }
    red[e] = (c * (1.0f / (float)N_TOKENS)) * (p * (1.0f / (float)N_TOKENS));
    __syncthreads();
    for (int off = 32; off > 0; off >>= 1) {
        if (e < off) red[e] += red[e + off];
        __syncthreads();
    }
    if (e == 0) out_aux[0] = 0.01f * red[0] * (float)N_EXP;
}

// ---------------------------------------------------------------------------
extern "C" void kernel_launch(void* const* d_in, const int* in_sizes, int n_in,
                              void* d_out, int out_size, void* d_ws, size_t ws_size,
                              hipStream_t stream) {
    const float* hidden = (const float*)d_in[0];   // (16384, 2048) fp32
    const float* gate_w = (const float*)d_in[1];   // (2048, 64)   fp32
    float* out = (float*)d_out;

    // workspace layout
    unsigned short* pb = (unsigned short*)d_ws;                     // 256 KB packed B
    size_t off = (size_t)(HID / 32) * 4 * 32 * 16 * sizeof(unsigned short);
    float* part = (float*)((char*)d_ws + off);                      // 16 MB partial logits
    off += (size_t)NTILES * KSPLIT * 16 * N_EXP * sizeof(float);
    float* ws_prob = (float*)((char*)d_ws + off);
    float* ws_cnt  = ws_prob + (size_t)NPBLOCKS * N_EXP;

    float* out_w   = out;                              // (16384, 8)
    float* out_i   = out + (size_t)N_TOKENS * TOPK;    // (16384, 8) indices as float
    float* out_aux = out + (size_t)2 * N_TOKENS * TOPK;

    pack_b_kernel<<<32, 256, 0, stream>>>(gate_w, pb);
    gemm_kernel<<<(NTILES / 8) * KSPLIT, 256, 0, stream>>>(hidden, pb, part);
    postproc_kernel<<<NPBLOCKS, 256, 0, stream>>>(part, out_w, out_i, ws_prob, ws_cnt);
    aux_kernel<<<1, 64, 0, stream>>>(ws_prob, ws_cnt, out_aux);
}
